// CrossAttention_87101936763185
// MI455X (gfx1250) — matrix operand order repaired and checked by reading, not compiled, not run on hardware
//
#include <hip/hip_runtime.h>
#include <math.h>

// ---------- problem constants ----------
#define DIMV      1024
#define CROSS_DIMV 768
#define NHEADS      16
#define HDIM        64
#define BATCH        4
#define SQ        2048
#define SK        1024
#define MQ        (BATCH * SQ)   // 8192
#define MK        (BATCH * SK)   // 4096

// 0.125 (=1/sqrt(HDIM)) * log2(e): moves softmax into exp2 domain
#define QSCALE 0.18033688011112042f

// ---------- types ----------
typedef __bf16 bf16_t;
typedef __attribute__((ext_vector_type(16))) __bf16 v16bf;
typedef __attribute__((ext_vector_type(8)))  __bf16 v8bf;
typedef __attribute__((ext_vector_type(4)))  __bf16 v4bf;
typedef __attribute__((ext_vector_type(8)))  float  v8f;

union AB16 { v16bf v; v8bf h[2]; };

__device__ __forceinline__ bf16_t f2bf(float f) {
  union { float f; unsigned int u; } a; a.f = f;
  unsigned int r = a.u + 0x7FFFu + ((a.u >> 16) & 1u);
  union { unsigned short s; bf16_t b; } o; o.s = (unsigned short)(r >> 16);
  return o.b;
}

#if __has_builtin(__builtin_amdgcn_exp2f)
#define EXP2F(x) __builtin_amdgcn_exp2f(x)
#else
#define EXP2F(x) exp2f(x)
#endif

// ---------- 16-lane half-wave butterfly reductions (pure-VALU DPP) ----------
#if __has_builtin(__builtin_amdgcn_update_dpp)
#define DPP_QUAD_XOR1 0xB1   // quad_perm [1,0,3,2]
#define DPP_QUAD_XOR2 0x4E   // quad_perm [2,3,0,1]
#define DPP_ROW_HMIRR 0x141  // row_half_mirror (xor4 once quads uniform)
#define DPP_ROW_MIRR  0x140  // row_mirror      (xor8 once 8-groups uniform)
template <int CTRL>
__device__ __forceinline__ float dpp_mov(float x) {
  int i = __builtin_bit_cast(int, x);
  int r = __builtin_amdgcn_update_dpp(i, i, CTRL, 0xF, 0xF, false);
  return __builtin_bit_cast(float, r);
}
__device__ __forceinline__ float redmax16(float x) {
  x = fmaxf(x, dpp_mov<DPP_QUAD_XOR1>(x));
  x = fmaxf(x, dpp_mov<DPP_QUAD_XOR2>(x));
  x = fmaxf(x, dpp_mov<DPP_ROW_HMIRR>(x));
  x = fmaxf(x, dpp_mov<DPP_ROW_MIRR>(x));
  return x;
}
__device__ __forceinline__ float redsum16(float x) {
  x += dpp_mov<DPP_QUAD_XOR1>(x);
  x += dpp_mov<DPP_QUAD_XOR2>(x);
  x += dpp_mov<DPP_ROW_HMIRR>(x);
  x += dpp_mov<DPP_ROW_MIRR>(x);
  return x;
}
#else
__device__ __forceinline__ float redmax16(float x) {
  x = fmaxf(x, __shfl_xor(x, 1, 32));
  x = fmaxf(x, __shfl_xor(x, 2, 32));
  x = fmaxf(x, __shfl_xor(x, 4, 32));
  x = fmaxf(x, __shfl_xor(x, 8, 32));
  return x;
}
__device__ __forceinline__ float redsum16(float x) {
  x += __shfl_xor(x, 1, 32);
  x += __shfl_xor(x, 2, 32);
  x += __shfl_xor(x, 4, 32);
  x += __shfl_xor(x, 8, 32);
  return x;
}
#endif

// A-matrix fragment, 16x32 bf16, row-major source with leading dim ld.
// lane l: row m = l&15; k-halves by l>>4 (ISA 7.12.2 16-bit A layout).
__device__ __forceinline__ v16bf load_a_frag(const bf16_t* base, int ld, int lane) {
  int m = lane & 15, g = lane >> 4;
  const bf16_t* p = base + (size_t)m * ld + g * 8;
  AB16 f;
  f.h[0] = *(const v8bf*)p;          // k = 8g .. 8g+7
  f.h[1] = *(const v8bf*)(p + 16);   // k = 16+8g .. 16+8g+7
  return f.v;
}

// B-matrix fragment, 32x16 bf16, from source laid out as [n][k] with leading
// dim ld (B[k][n] = src[n*ld + k]). lane l: col n = l&15; k-halves by l>>4:
// k = 8g..8g+7 then 16+8g..16+8g+7 (same order as load_a_frag).
__device__ __forceinline__ v16bf load_b_frag(const bf16_t* base, int ld, int lane) {
  int n = lane & 15, g = lane >> 4;
  const bf16_t* p = base + (size_t)n * ld + g * 8;
  AB16 f;
  f.h[0] = *(const v8bf*)p;
  f.h[1] = *(const v8bf*)(p + 16);
  return f.v;
}

__device__ __forceinline__ v8f wmma_bf16(v16bf a, v16bf b, v8f c) {
  return __builtin_amdgcn_wmma_f32_16x16x32_bf16(
      false, a, false, b, (short)0, c, false, false);
}

// ---------- fp32 -> bf16 conversion (vectorized x4, row-major kept) ----------
__global__ void cvt_bf16_kernel(const float* __restrict__ in,
                                bf16_t* __restrict__ out, int n4) {
  int i = blockIdx.x * blockDim.x + threadIdx.x;
  if (i >= n4) return;
  float4 f = ((const float4*)in)[i];
  v4bf o;
  o[0] = f2bf(f.x); o[1] = f2bf(f.y); o[2] = f2bf(f.z); o[3] = f2bf(f.w);
  ((v4bf*)out)[i] = o;
}

// ---------- fp32 [K][N] -> bf16 [N][K] transposing conversion ----------
// 32x32 tiles through LDS, coalesced on both sides. Grid (N/32, K/32),
// block (32,8).
__global__ __launch_bounds__(256)
void cvt_transpose_kernel(const float* __restrict__ in, bf16_t* __restrict__ out,
                          int K, int N) {
  __shared__ float tile[32][33];
  int k0 = blockIdx.y * 32;
  int n0 = blockIdx.x * 32;
  int tx = threadIdx.x;   // 0..31
  int ty = threadIdx.y;   // 0..7
#pragma unroll
  for (int j = 0; j < 4; ++j)
    tile[ty + j * 8][tx] = in[(size_t)(k0 + ty + j * 8) * N + n0 + tx];
  __syncthreads();
#pragma unroll
  for (int j = 0; j < 4; ++j)
    out[(size_t)(n0 + ty + j * 8) * K + k0 + tx] = f2bf(tile[tx][ty + j * 8]);
}

// ---------- bf16 GEMM, no LDS: C[M,N] = A[M,K] @ Wt[N,K]^T + bias ----------
// Weights arrive pre-transposed ([N][K]) so both A and B fragments are
// contiguous global b128 loads; no staging, no barriers. 128 threads =
// 4 waves; block tile 64x128; each wave: 16 rows x 128 cols (8 WMMA/k-step,
// one A fragment feeds all 8).
// MODE: 0 = bf16 row-major out, 1 = f32 row-major out,
//       2 = bf16 out in V-transposed layout [b,h][dim(64)][key(SK)]
template <int MODE>
__global__ __launch_bounds__(128)
void gemm_bf16_kernel(const bf16_t* __restrict__ A, const bf16_t* __restrict__ Wt,
                      const float* __restrict__ bias, void* __restrict__ out,
                      float oscale, int M, int N, int K) {
  const int tn0 = blockIdx.x * 128;
  const int tm0 = blockIdx.y * 64;
  const int wave = threadIdx.x >> 5;
  const int lane = threadIdx.x & 31;

  v8f zero = {0.f, 0.f, 0.f, 0.f, 0.f, 0.f, 0.f, 0.f};
  v8f acc[8] = {zero, zero, zero, zero, zero, zero, zero, zero};

  const bf16_t* Arow = A + (size_t)(tm0 + wave * 16) * K;

  for (int kb = 0; kb < K; kb += 32) {
    if (kb + 32 < K)
      __builtin_prefetch(Arow + (size_t)(lane & 15) * K + kb + 32, 0, 3);
    v16bf af = load_a_frag(Arow + kb, K, lane);
#pragma unroll
    for (int nt = 0; nt < 8; ++nt) {
      const bf16_t* wp = Wt + (size_t)(tn0 + nt * 16) * K + kb;
      acc[nt] = wmma_bf16(af, load_b_frag(wp, K, lane), acc[nt]);
    }
  }

  const int n = lane & 15, g = lane >> 4;
#pragma unroll
  for (int nt = 0; nt < 8; ++nt) {
    int col = tn0 + nt * 16 + n;
    float bv = bias[col];
#pragma unroll
    for (int r = 0; r < 8; ++r) {
      int row = tm0 + wave * 16 + r + 8 * g;
      float v = (acc[nt][r] + bv) * oscale;
      if (MODE == 1) {
        ((float*)out)[(size_t)row * N + col] = v;
      } else if (MODE == 0) {
        ((bf16_t*)out)[(size_t)row * N + col] = f2bf(v);
      } else { // MODE == 2: V-transposed [ (b*NHEADS+h)*HDIM + d ][ key ]
        int bb  = row >> 10;        // row / SK
        int key = row & (SK - 1);
        int hh  = col >> 6;         // col / HDIM
        int dd  = col & (HDIM - 1);
        ((bf16_t*)out)[(((size_t)bb * NHEADS + hh) * HDIM + dd) * SK + key] =
            f2bf(v);
      }
    }
  }
}

// ---------- flash attention ----------
// grid: B * NHEADS * (SQ/64) blocks, 128 threads (4 independent waves).
// Q comes pre-scaled by 0.125*log2e, so softmax is pure exp2. 64-key chunks:
// one online-softmax update (DPP trees, corr rescale) per 64 keys. Fully
// masked chunks are skipped (mask is monotonic: arange >= length). K reads
// row-major global; V reads the pre-transposed [b,h][dim][key] layout; only
// the wave-private P lane-transpose touches LDS; no __syncthreads anywhere.
__global__ __launch_bounds__(128)
void attn_kernel(const bf16_t* __restrict__ Qb, const bf16_t* __restrict__ Kb,
                 const bf16_t* __restrict__ Vt,
                 const unsigned char* __restrict__ mask,
                 bf16_t* __restrict__ ctx) {
  __shared__ __align__(16) bf16_t Plds[4][16 * 64]; // per-wave [row][key64]

  const int QT  = SQ / 64; // 32
  int bid = blockIdx.x;
  int qt = bid % QT;
  int h  = (bid / QT) % NHEADS;
  int b  = bid / (QT * NHEADS);

  const int wave = threadIdx.x >> 5;
  const int lane = threadIdx.x & 31;
  const int n = lane & 15, g = lane >> 4;

  const bf16_t* Kbase = Kb + (size_t)b * SK * DIMV + h * HDIM;        // ld=DIMV
  const bf16_t* Vbase = Vt + ((size_t)b * NHEADS + h) * HDIM * SK;    // ld=SK
  const unsigned char* mbase = mask + (size_t)b * SK;

  // Q fragments for this wave's 16 rows (head-dim 64 = 2 k-steps of 32)
  const size_t qrow0 = (size_t)(b * SQ + qt * 64 + wave * 16);
  v16bf qf[2];
#pragma unroll
  for (int kd = 0; kd < 2; ++kd)
    qf[kd] = load_a_frag(Qb + qrow0 * DIMV + h * HDIM + kd * 32, DIMV, lane);

  v8f zero = {0.f, 0.f, 0.f, 0.f, 0.f, 0.f, 0.f, 0.f};
  v8f oacc[4] = {zero, zero, zero, zero};
  float mrow[8], lsum[8];
#pragma unroll
  for (int r = 0; r < 8; ++r) { mrow[r] = -__builtin_inff(); lsum[r] = 0.f; }

  const int NC = SK / 64; // 16 chunks of 64 keys
#pragma unroll 1
  for (int c = 0; c < NC; ++c) {
    const int key0 = c * 64;

    // monotonic key-padding mask: first key masked => whole chunk masked,
    // contributes exactly zero. Uniform branch across the block.
    if (mbase[key0]) continue;

    // prefetch next chunk of K and V (global_prefetch_b8)
    if (c + 1 < NC) {
      __builtin_prefetch(Kbase + (size_t)(key0 + 64 + lane) * DIMV, 0, 3);
      __builtin_prefetch(Kbase + (size_t)(key0 + 96 + lane) * DIMV, 0, 3);
      __builtin_prefetch(Vbase + (size_t)(lane * 2) * SK + key0 + 64, 0, 3);
    }

    // scores (pre-scaled, exp2 domain): four 16x16 key tiles, 2 WMMA each;
    // kd outer so same-accumulator WMMAs are 4 apart (RAW hazard distance).
    v8f sc[4] = {zero, zero, zero, zero};
#pragma unroll
    for (int kd = 0; kd < 2; ++kd)
#pragma unroll
      for (int t = 0; t < 4; ++t) {
        const bf16_t* kp = Kbase + (size_t)(key0 + t * 16) * DIMV + kd * 32;
        sc[t] = wmma_bf16(qf[kd], load_b_frag(kp, DIMV, lane), sc[t]);
      }

    // mask + one online-softmax update per 64 keys (DPP half-wave trees);
    // P written to wave-private LDS inline (C layout -> A layout transpose)
    unsigned char mk[4];
#pragma unroll
    for (int t = 0; t < 4; ++t) mk[t] = mbase[key0 + t * 16 + n];
    bf16_t* pl = Plds[wave];
#pragma unroll
    for (int r = 0; r < 8; ++r) {
      float s[4];
#pragma unroll
      for (int t = 0; t < 4; ++t)
        s[t] = mk[t] ? -__builtin_inff() : sc[t][r];
      float cm = redmax16(fmaxf(fmaxf(s[0], s[1]), fmaxf(s[2], s[3])));
      float newm = fmaxf(mrow[r], cm);
      float corr = EXP2F(mrow[r] - newm);
      float psum = 0.f;
      int rowoff = (r + 8 * g) * 64;
#pragma unroll
      for (int t = 0; t < 4; ++t) {
        float p = EXP2F(s[t] - newm);
        psum += p;
        pl[rowoff + t * 16 + n] = f2bf(p);
      }
      float rs = redsum16(psum);
      lsum[r] = lsum[r] * corr + rs;
      mrow[r] = newm;
#pragma unroll
      for (int ot = 0; ot < 4; ++ot) oacc[ot][r] *= corr;
    }
    __builtin_amdgcn_wave_barrier();  // same-wave DS ops are in-order

    // O += P @ V  (A = P 16x64 as two 16x32 frags; B from Vt[dim][key]);
    // ot inner so same-accumulator WMMAs are 4 apart.
#pragma unroll
    for (int kd2 = 0; kd2 < 2; ++kd2) {
      v16bf pf = load_a_frag(pl + kd2 * 32, 64, lane);
#pragma unroll
      for (int ot = 0; ot < 4; ++ot) {
        const bf16_t* vp = Vbase + (size_t)(ot * 16) * SK + key0 + kd2 * 32;
        oacc[ot] = wmma_bf16(pf, load_b_frag(vp, SK, lane), oacc[ot]);
      }
    }
  }

  // normalize and store context (bf16) at this head's columns
  float rinv[8];
#pragma unroll
  for (int r = 0; r < 8; ++r) rinv[r] = 1.0f / lsum[r];
#pragma unroll
  for (int ot = 0; ot < 4; ++ot) {
    int col = h * HDIM + ot * 16 + n;
#pragma unroll
    for (int r = 0; r < 8; ++r) {
      size_t row = qrow0 + r + 8 * g;
      ctx[row * DIMV + col] = f2bf(oacc[ot][r] * rinv[r]);
    }
  }
}

// ---------- host launcher ----------
extern "C" void kernel_launch(void* const* d_in, const int* in_sizes, int n_in,
                              void* d_out, int out_size, void* d_ws, size_t ws_size,
                              hipStream_t stream) {
  const float* x  = (const float*)d_in[0];
  const float* cx = (const float*)d_in[1];
  const unsigned char* mask = (const unsigned char*)d_in[2];
  const float* Wq = (const float*)d_in[3];
  const float* bq = (const float*)d_in[4];
  const float* Wk = (const float*)d_in[5];
  const float* bk = (const float*)d_in[6];
  const float* Wv = (const float*)d_in[7];
  const float* bv = (const float*)d_in[8];
  const float* Wo = (const float*)d_in[9];
  const float* bo = (const float*)d_in[10];
  float* out = (float*)d_out;

  char* ws = (char*)d_ws;
  size_t off = 0;
  auto carve = [&](size_t elems) {
    bf16_t* p = (bf16_t*)(ws + off);
    off += ((elems * sizeof(bf16_t)) + 255) & ~(size_t)255;
    return p;
  };
  bf16_t* xb   = carve((size_t)MQ * DIMV);
  bf16_t* cxb  = carve((size_t)MK * CROSS_DIMV);
  bf16_t* wqt  = carve((size_t)DIMV * DIMV);        // [N][K] transposed
  bf16_t* wkt  = carve((size_t)CROSS_DIMV * DIMV);
  bf16_t* wvt  = carve((size_t)CROSS_DIMV * DIMV);
  bf16_t* wot  = carve((size_t)DIMV * DIMV);
  bf16_t* qb   = carve((size_t)MQ * DIMV);
  bf16_t* kb   = carve((size_t)MK * DIMV);
  bf16_t* vtb  = carve((size_t)MK * DIMV);          // V-transposed [b,h][dim][key]
  bf16_t* ctxb = carve((size_t)MQ * DIMV);

  auto cvt = [&](const float* src, bf16_t* dst, size_t elems) {
    int n4 = (int)(elems / 4);
    cvt_bf16_kernel<<<(n4 + 255) / 256, 256, 0, stream>>>(src, dst, n4);
  };
  cvt(x,  xb,  (size_t)MQ * DIMV);
  cvt(cx, cxb, (size_t)MK * CROSS_DIMV);

  auto cvtT = [&](const float* src, bf16_t* dst, int K, int N) {
    cvt_transpose_kernel<<<dim3(N / 32, K / 32), dim3(32, 8), 0, stream>>>(
        src, dst, K, N);
  };
  cvtT(Wq, wqt, DIMV, DIMV);
  cvtT(Wk, wkt, CROSS_DIMV, DIMV);
  cvtT(Wv, wvt, CROSS_DIMV, DIMV);
  cvtT(Wo, wot, DIMV, DIMV);

  // Q = (x @ Wq + bq) * 0.125*log2e   (bf16, row-major, pre-scaled)
  gemm_bf16_kernel<0><<<dim3(DIMV / 128, MQ / 64), 128, 0, stream>>>(
      xb, wqt, bq, qb, QSCALE, MQ, DIMV, DIMV);
  // K = cx @ Wk + bk                  (bf16, row-major)
  gemm_bf16_kernel<0><<<dim3(DIMV / 128, MK / 64), 128, 0, stream>>>(
      cxb, wkt, bk, kb, 1.0f, MK, DIMV, CROSS_DIMV);
  // V = cx @ Wv + bv                  (bf16, transposed per-head layout)
  gemm_bf16_kernel<2><<<dim3(DIMV / 128, MK / 64), 128, 0, stream>>>(
      cxb, wvt, bv, vtb, 1.0f, MK, DIMV, CROSS_DIMV);

  // flash attention -> ctx (bf16)
  attn_kernel<<<BATCH * NHEADS * (SQ / 64), 128, 0, stream>>>(
      qb, kb, vtb, mask, ctxb);

  // out = ctx @ Wo + bo               (fp32)
  gemm_bf16_kernel<1><<<dim3(DIMV / 128, MQ / 64), 128, 0, stream>>>(
      ctxb, wot, bo, out, 1.0f, MQ, DIMV, DIMV);
}
